// SelfAttention_68006512164897
// MI455X (gfx1250) — compile-verified, matrix operand order checked by reading, not run on hardware
//
#include <hip/hip_runtime.h>
#include <hip/hip_bf16.h>

// Problem constants (fixed by the reference)
#define B_SZ   4
#define S_LEN  2048
#define E_DIM  1024
#define H_NUM  16
#define D_HEAD 64

typedef __bf16 bf16;
typedef __attribute__((ext_vector_type(16))) __bf16 v16bf;
typedef __attribute__((ext_vector_type(8)))  __bf16 bf16x8;
typedef __attribute__((ext_vector_type(8)))  float  v8f;
typedef __attribute__((ext_vector_type(4)))  int    v4i;

// ---- CDNA5 async global->LDS copy (ASYNCcnt path, ISA 08_async_tensor §4) ----
#if __has_builtin(__builtin_amdgcn_global_load_async_to_lds_b128)
#define HAS_ASYNC 1
#else
#define HAS_ASYNC 0
#endif

// b128 async copy moves one 16-byte packet per lane; builtin wants
// v4i pointers: AS1 (global) source, AS3 (LDS) destination.
typedef __attribute__((address_space(1))) v4i gv4i;
typedef __attribute__((address_space(3))) v4i lv4i;

__device__ __forceinline__ void async_copy_b128(const bf16* g, bf16* l) {
#if HAS_ASYNC
  // integer round-trip: flat-addr low 32 bits == LDS offset (aperture rule),
  // AS3 pointers are 32-bit
  __builtin_amdgcn_global_load_async_to_lds_b128(
      (gv4i*)(size_t)g, (lv4i*)(unsigned)(size_t)l, 0, 0);
#else
  *(bf16x8*)l = *(const bf16x8*)g;
#endif
}

__device__ __forceinline__ void async_wait() {
#if HAS_ASYNC
#if __has_builtin(__builtin_amdgcn_s_wait_asynccnt)
  __builtin_amdgcn_s_wait_asynccnt(0);
#else
  asm volatile("s_wait_asynccnt 0x0" ::: "memory");
#endif
#endif
}

__device__ __forceinline__ v8f v8f_zero() {
  v8f z = {0.f,0.f,0.f,0.f,0.f,0.f,0.f,0.f};
  return z;
}

// D = A(16x32) * B(32x16) + C, bf16 in / f32 accum
__device__ __forceinline__ v8f wmma_bf16(v16bf a, v16bf b, v8f c) {
  return __builtin_amdgcn_wmma_f32_16x16x32_bf16(
      /*neg_a=*/false, a, /*neg_b=*/false, b,
      /*c_mod=*/(short)0, c, /*reuse_a=*/false, /*reuse_b=*/false);
}

// Load one 16x32 bf16 operand fragment (A row-major, or B via its transposed
// row-major image -- the CDNA5 VGPR layouts for 16-bit A 16x32 and B 32x16 are
// symmetric). lane&15 selects the row(A)/col(B); K is packed as
// [kb..kb+7, kb+16..kb+23] with kb = (lane&16)?8:0 -> two 16B ds loads.
__device__ __forceinline__ v16bf load_frag(const bf16* base, int ld, int lane) {
  const int r  = lane & 15;
  const int kb = (lane & 16) ? 8 : 0;
  const bf16* p = base + r * ld + kb;
  bf16x8 lo = *(const bf16x8*)(p);
  bf16x8 hi = *(const bf16x8*)(p + 16);
  return __builtin_shufflevector(lo, hi, 0,1,2,3,4,5,6,7,8,9,10,11,12,13,14,15);
}

// ---------------- conversion kernels ----------------

__global__ __launch_bounds__(256) void cvt_f32_bf16_kernel(
    const float* __restrict__ in, bf16* __restrict__ out, int n) {
  int i = (blockIdx.x * 256 + threadIdx.x) * 8;
  if (i >= n) return;
  const float4* p = (const float4*)(in + i);
  float4 a = p[0], b = p[1];
  bf16x8 o;
  o[0]=(bf16)a.x; o[1]=(bf16)a.y; o[2]=(bf16)a.z; o[3]=(bf16)a.w;
  o[4]=(bf16)b.x; o[5]=(bf16)b.y; o[6]=(bf16)b.z; o[7]=(bf16)b.w;
  *(bf16x8*)(out + i) = o;
}

// W [in=E][out=E] fp32 -> Wt [out][in] bf16 (so GEMM B-operand frags are
// plain row-major loads)
__global__ __launch_bounds__(256) void transpose_w_kernel(
    const float* __restrict__ w, bf16* __restrict__ wt) {
  int idx  = blockIdx.x * 256 + threadIdx.x;  // E*E threads
  int nOut = idx & (E_DIM - 1);
  int kIn  = idx >> 10;
  wt[(size_t)nOut * E_DIM + kIn] = (bf16)w[(size_t)kIn * E_DIM + nOut];
}

// ---------------- tiled bf16 WMMA GEMM (double-buffered async LDS) ----------
// C[M,N] = A[M,K] * Bt[N,K]^T + bias, 128x128 block tile, 8 waves,
// each wave computes 32x64 (2x4 wmma frags), K-step 32.
// MODE 0: bf16 out, head-major  [B,H,S,D]   (Q, K)
// MODE 1: bf16 out, head-transp [B,H,D,S]   (V)
// MODE 2: fp32 out, flat        [M,N]       (final projection)

#define GT_K 32
#define LDP  (GT_K + 8)   // 40 bf16 = 80B row stride (16B-aligned)

template<int MODE>
__global__ __launch_bounds__(256) void gemm_bf16_kernel(
    const bf16*  __restrict__ A,
    const bf16*  __restrict__ Bt,
    const float* __restrict__ bias,
    void*        __restrict__ Cout,
    int M, int N, int K)
{
  __shared__ bf16 As[2][128][LDP];
  __shared__ bf16 Bs[2][128][LDP];

  const int tid  = threadIdx.x;
  const int lane = tid & 31;
  const int wid  = tid >> 5;            // 0..7
  const int wm   = (wid & 3) * 32;      // wave M offset inside tile
  const int wn   = (wid >> 2) * 64;     // wave N offset inside tile

  const int mBase = blockIdx.y * 128;
  const int nBase = blockIdx.x * 128;

  const int ldRow = tid >> 1;           // 0..127
  const int ldSeg = (tid & 1) * 16;     // 0 or 16 (bf16 elems)

  // stage one 128x32 tile of A and Bt into LDS buffer `buf` (async)
  auto stage = [&](int buf, int k0) {
    const bf16* ga = A  + (size_t)(mBase + ldRow) * K + k0 + ldSeg;
    const bf16* gb = Bt + (size_t)(nBase + ldRow) * K + k0 + ldSeg;
    async_copy_b128(ga,     &As[buf][ldRow][ldSeg]);
    async_copy_b128(ga + 8, &As[buf][ldRow][ldSeg + 8]);
    async_copy_b128(gb,     &Bs[buf][ldRow][ldSeg]);
    async_copy_b128(gb + 8, &Bs[buf][ldRow][ldSeg + 8]);
  };

  v8f acc[2][4];
  #pragma unroll
  for (int i = 0; i < 2; ++i)
    #pragma unroll
    for (int j = 0; j < 4; ++j) acc[i][j] = v8f_zero();

  stage(0, 0);
  async_wait();
  __syncthreads();

  for (int k0 = 0; k0 < K; k0 += GT_K) {
    const int cur = (k0 / GT_K) & 1;
    if (k0 + GT_K < K) stage(cur ^ 1, k0 + GT_K);   // overlap next tile

    v16bf af[2], bfv[4];
    #pragma unroll
    for (int i = 0; i < 2; ++i) af[i]  = load_frag(&As[cur][wm + i*16][0], LDP, lane);
    #pragma unroll
    for (int j = 0; j < 4; ++j) bfv[j] = load_frag(&Bs[cur][wn + j*16][0], LDP, lane);
    #pragma unroll
    for (int i = 0; i < 2; ++i)
      #pragma unroll
      for (int j = 0; j < 4; ++j)
        acc[i][j] = wmma_bf16(af[i], bfv[j], acc[i][j]);

    async_wait();
    __syncthreads();
  }

  // epilogue: C-frag layout -> lane holds col = lane&15, rows rOff..rOff+7
  const int rOff = (lane & 16) ? 8 : 0;
  const int cl   = lane & 15;
  #pragma unroll
  for (int i = 0; i < 2; ++i) {
    #pragma unroll
    for (int j = 0; j < 4; ++j) {
      const int col = nBase + wn + j*16 + cl;
      const float bv = bias[col];
      #pragma unroll
      for (int r = 0; r < 8; ++r) {
        const int row = mBase + wm + i*16 + rOff + r;
        const float val = acc[i][j][r] + bv;
        if (MODE == 0) {
          const int b = row >> 11, s = row & (S_LEN - 1);
          const int h = col >> 6,  d = col & (D_HEAD - 1);
          ((bf16*)Cout)[(((size_t)(b*H_NUM + h) * S_LEN) + s) * D_HEAD + d] = (bf16)val;
        } else if (MODE == 1) {
          const int b = row >> 11, s = row & (S_LEN - 1);
          const int h = col >> 6,  d = col & (D_HEAD - 1);
          ((bf16*)Cout)[(((size_t)(b*H_NUM + h) * D_HEAD) + d) * S_LEN + s] = (bf16)val;
        } else {
          ((float*)Cout)[(size_t)row * N + col] = val;
        }
      }
    }
  }
}

// ---------------- flash attention (online softmax, async double buffer) -----
// grid = (S/64, B*H), block = 128 (4 waves). Wave w owns queries
// qBase..qBase+15. K tiles of 64 keys staged async into LDS; V comes
// pre-transposed [B,H,D,S] so its B-operand frags are plain row-major loads.
// P is bounced through LDS to re-shape C-layout -> A-layout for PV.

__global__ __launch_bounds__(128) void flash_attn_kernel(
    const bf16* __restrict__ Q,    // [B*H, S, D]
    const bf16* __restrict__ Kh,   // [B*H, S, D]
    const bf16* __restrict__ Vt,   // [B*H, D, S]
    bf16*       __restrict__ Out)  // [B*S, E]
{
  __shared__ bf16 Ks[2][64][72];     // [key][d]
  __shared__ bf16 Vs[2][64][72];     // [d][key]
  __shared__ bf16 Ps[4][16][72];     // per-wave P tile [q][key]

  const int tid   = threadIdx.x;
  const int lane  = tid & 31;
  const int w     = tid >> 5;                    // wave 0..3
  const int bh    = blockIdx.y;                  // b*H + h
  const int qBase = blockIdx.x * 64 + w * 16;

  const bf16* qhead = Q  + (size_t)bh * S_LEN * D_HEAD;
  const bf16* khead = Kh + (size_t)bh * S_LEN * D_HEAD;
  const bf16* vhead = Vt + (size_t)bh * D_HEAD * S_LEN;

  const int r    = lane & 15;
  const int kb   = (lane & 16) ? 8 : 0;
  const int rOff = (lane & 16) ? 8 : 0;
  const int cl   = lane & 15;

  const int ldRow = tid >> 1;          // 0..63
  const int ldSeg = (tid & 1) * 32;    // 0 or 32 (bf16 elems)

  // stage K tile [64 keys x 64 d] and V^T tile [64 d x 64 keys] (async)
  auto stage = [&](int buf, int kt) {
    const bf16* gk = khead + (size_t)(kt + ldRow) * D_HEAD + ldSeg;
    const bf16* gv = vhead + (size_t)ldRow * S_LEN + kt + ldSeg;
    #pragma unroll
    for (int u = 0; u < 4; ++u) {
      async_copy_b128(gk + u*8, &Ks[buf][ldRow][ldSeg + u*8]);
      async_copy_b128(gv + u*8, &Vs[buf][ldRow][ldSeg + u*8]);
    }
  };

  // Q A-frags: 16 rows x 64 d -> 2 frags, loaded straight from global
  v16bf qf[2];
  #pragma unroll
  for (int j = 0; j < 2; ++j) {
    const bf16* p = qhead + (size_t)(qBase + r) * D_HEAD + j*32 + kb;
    bf16x8 lo = *(const bf16x8*)(p);
    bf16x8 hi = *(const bf16x8*)(p + 16);
    qf[j] = __builtin_shufflevector(lo, hi, 0,1,2,3,4,5,6,7,8,9,10,11,12,13,14,15);
  }

  float mrow[8], lrow[8];
  v8f o[4];
  #pragma unroll
  for (int i = 0; i < 8; ++i) { mrow[i] = -1e30f; lrow[i] = 0.f; }
  #pragma unroll
  for (int j = 0; j < 4; ++j) o[j] = v8f_zero();

  stage(0, 0);
  async_wait();
  __syncthreads();

  for (int kt = 0; kt < S_LEN; kt += 64) {
    const int cur = (kt >> 6) & 1;
    if (kt + 64 < S_LEN) stage(cur ^ 1, kt + 64);   // overlap next tile

    // S = Q K^T / sqrt(D): 4 key-subtiles of 16, contraction D=64 = 2 wmma each
    v8f sfr[4];
    #pragma unroll
    for (int nt = 0; nt < 4; ++nt) {
      v8f s = v8f_zero();
      s = wmma_bf16(qf[0], load_frag(&Ks[cur][nt*16][0],  72, lane), s);
      s = wmma_bf16(qf[1], load_frag(&Ks[cur][nt*16][32], 72, lane), s);
      sfr[nt] = s * 0.125f;   // 1/sqrt(64)
    }

    // online softmax: 8 rows per lane; row reduce across the 16 lanes of the
    // half-wave (C-layout: lanes 0-15 = rows 0-7, lanes 16-31 = rows 8-15)
    float pnew[4][8];
    #pragma unroll
    for (int i = 0; i < 8; ++i) {
      float mx = sfr[0][i];
      #pragma unroll
      for (int nt = 1; nt < 4; ++nt) mx = fmaxf(mx, sfr[nt][i]);
      #pragma unroll
      for (int off = 8; off >= 1; off >>= 1) mx = fmaxf(mx, __shfl_xor(mx, off, 32));
      const float mnew  = fmaxf(mrow[i], mx);
      const float alpha = __expf(mrow[i] - mnew);
      float rsum = 0.f;
      #pragma unroll
      for (int nt = 0; nt < 4; ++nt) {
        const float p = __expf(sfr[nt][i] - mnew);
        pnew[nt][i] = p;
        rsum += p;
      }
      #pragma unroll
      for (int off = 8; off >= 1; off >>= 1) rsum += __shfl_xor(rsum, off, 32);
      lrow[i] = lrow[i] * alpha + rsum;
      mrow[i] = mnew;
      #pragma unroll
      for (int nt = 0; nt < 4; ++nt) o[nt][i] = o[nt][i] * alpha;
    }

    // bounce P through LDS: C-layout -> A-layout
    #pragma unroll
    for (int nt = 0; nt < 4; ++nt)
      #pragma unroll
      for (int i = 0; i < 8; ++i)
        Ps[w][rOff + i][nt*16 + cl] = (bf16)pnew[nt][i];

    // O += P @ V  (contraction over 64 keys = 2 wmma per output d-subtile)
    v16bf pf0 = load_frag(&Ps[w][0][0],  72, lane);
    v16bf pf1 = load_frag(&Ps[w][0][32], 72, lane);
    #pragma unroll
    for (int nt = 0; nt < 4; ++nt) {
      o[nt] = wmma_bf16(pf0, load_frag(&Vs[cur][nt*16][0],  72, lane), o[nt]);
      o[nt] = wmma_bf16(pf1, load_frag(&Vs[cur][nt*16][32], 72, lane), o[nt]);
    }

    async_wait();
    __syncthreads();
  }

  // normalize and store bf16 flat [B*S, E] for the final projection
  const int b = bh >> 4;
  const int h = bh & (H_NUM - 1);
  #pragma unroll
  for (int i = 0; i < 8; ++i) {
    const float inv  = 1.0f / lrow[i];
    const int   grow = b * S_LEN + qBase + rOff + i;
    #pragma unroll
    for (int nt = 0; nt < 4; ++nt) {
      const int col = h * D_HEAD + nt*16 + cl;
      Out[(size_t)grow * E_DIM + col] = (bf16)(o[nt][i] * inv);
    }
  }
}

// ---------------- launch ----------------

extern "C" void kernel_launch(void* const* d_in, const int* in_sizes, int n_in,
                              void* d_out, int out_size, void* d_ws, size_t ws_size,
                              hipStream_t stream) {
  (void)in_sizes; (void)n_in; (void)out_size; (void)ws_size;
  const float* x  = (const float*)d_in[0];
  const float* Wq = (const float*)d_in[1];
  const float* bq = (const float*)d_in[2];
  const float* Wk = (const float*)d_in[3];
  const float* bk = (const float*)d_in[4];
  const float* Wv = (const float*)d_in[5];
  const float* bv = (const float*)d_in[6];
  const float* Wo = (const float*)d_in[7];
  const float* bo = (const float*)d_in[8];

  const size_t XB = (size_t)B_SZ * S_LEN * E_DIM;   // 8,388,608 elems
  const size_t WE = (size_t)E_DIM * E_DIM;          // 1,048,576 elems

  // workspace layout (bf16 elems): ~92 MB total
  bf16* xb   = (bf16*)d_ws;
  bf16* wqt  = xb   + XB;
  bf16* wkt  = wqt  + WE;
  bf16* wvt  = wkt  + WE;
  bf16* wot  = wvt  + WE;
  bf16* qb   = wot  + WE;   // [B,H,S,D]
  bf16* kbuf = qb   + XB;   // [B,H,S,D]
  bf16* vtb  = kbuf + XB;   // [B,H,D,S]
  bf16* ab   = vtb  + XB;   // [B*S,E] attention output

  const int M = B_SZ * S_LEN;   // 8192

  cvt_f32_bf16_kernel<<<(int)(XB / 8 / 256), 256, 0, stream>>>(x, xb, (int)XB);
  transpose_w_kernel<<<(int)(WE / 256), 256, 0, stream>>>(Wq, wqt);
  transpose_w_kernel<<<(int)(WE / 256), 256, 0, stream>>>(Wk, wkt);
  transpose_w_kernel<<<(int)(WE / 256), 256, 0, stream>>>(Wv, wvt);
  transpose_w_kernel<<<(int)(WE / 256), 256, 0, stream>>>(Wo, wot);

  dim3 ggrid(E_DIM / 128, M / 128);   // (8, 64)
  gemm_bf16_kernel<0><<<ggrid, 256, 0, stream>>>(xb, wqt, bq, qb,   M, E_DIM, E_DIM);
  gemm_bf16_kernel<0><<<ggrid, 256, 0, stream>>>(xb, wkt, bk, kbuf, M, E_DIM, E_DIM);
  gemm_bf16_kernel<1><<<ggrid, 256, 0, stream>>>(xb, wvt, bv, vtb,  M, E_DIM, E_DIM);

  flash_attn_kernel<<<dim3(S_LEN / 64, B_SZ * H_NUM), 128, 0, stream>>>(qb, kbuf, vtb, ab);

  gemm_bf16_kernel<2><<<ggrid, 256, 0, stream>>>(ab, wot, bo, d_out, M, E_DIM, E_DIM);
}
